// MoELayer_2276332667279
// MI455X (gfx1250) — matrix operand order repaired and checked by reading, not compiled, not run on hardware
//
#include <hip/hip_runtime.h>
#include <stdint.h>
#include <math.h>

// Problem constants (from reference): B*S=8192 tokens, H=1024, F=4096, E=8, top_k=2
#define T_TOK 8192
#define H_DIM 1024
#define F_DIM 4096
#define NEXP  8
#define M_TILE 32            // tokens per FFN block
#define FC 64                // F-chunk per fused iteration
#define XS_STRIDE 1040       // bf16 elems per Xs row (pad, 16B aligned rows: 2080B)
#define HS_STRIDE 72         // bf16 elems per Hs row (pad, 16B aligned rows: 144B)
#define FRAG_ELEMS 512       // one 16(N)x32(K) bf16 B-fragment: 32 lanes x 16 bf16

typedef __bf16 bf16_t;
typedef bf16_t v16bf __attribute__((ext_vector_type(16)));
typedef float  v8f   __attribute__((ext_vector_type(8)));
typedef unsigned int v4u __attribute__((ext_vector_type(4)));

// types for the async global->LDS builtin (param is vector_size(16) int, AS1/AS3)
typedef int vsi4 __attribute__((vector_size(16)));
typedef __attribute__((address_space(1))) vsi4* gv4p;
typedef __attribute__((address_space(3))) vsi4* lv4p;

union Frag { v16bf v; v4u q[2]; unsigned short s[16]; };

#if defined(__has_builtin)
#if __has_builtin(__builtin_amdgcn_global_load_async_to_lds_b128)
#define HAVE_ASYNC_LDS 1
#endif
#endif

__device__ __forceinline__ unsigned short f2bf(float f) {
  unsigned int u = __float_as_uint(f);
  unsigned int r = ((u >> 16) & 1u) + 0x7FFFu;   // round-to-nearest-even
  return (unsigned short)((u + r) >> 16);
}

__device__ __forceinline__ float gelu_tanh(float x) {
  float x3 = x * x * x;
  float t = tanhf(0.7978845608028654f * (x + 0.044715f * x3));
  return 0.5f * x * (1.0f + t);
}

__device__ __forceinline__ v8f wmma_bf16(v16bf a, v16bf b, v8f c) {
  // (neg_a, A, neg_b, B, c_mod, C, reuse_a, reuse_b)
  return __builtin_amdgcn_wmma_f32_16x16x32_bf16(false, a, false, b, (short)0, c,
                                                 false, false);
}

// ---------------------------------------------------------------- zero counters
__global__ void zero_cnt_kernel(int* __restrict__ cnt) {
  if (threadIdx.x < NEXP) cnt[threadIdx.x] = 0;
}

// ---------------------------------------------------------------- router
__global__ __launch_bounds__(256)
void router_kernel(const float* __restrict__ x, const float* __restrict__ rw,
                   const float* __restrict__ rb, int* __restrict__ cnt,
                   int* __restrict__ ids, float* __restrict__ awts)
{
  int t = blockIdx.x * 256 + threadIdx.x;
  if (t >= T_TOK) return;
  float acc[NEXP];
#pragma unroll
  for (int e = 0; e < NEXP; ++e) acc[e] = rb[e];
  const float4* xr = reinterpret_cast<const float4*>(x + (size_t)t * H_DIM);
  for (int h4 = 0; h4 < H_DIM / 4; ++h4) {
    float4 xv = xr[h4];
    float xs[4] = {xv.x, xv.y, xv.z, xv.w};
#pragma unroll
    for (int j = 0; j < 4; ++j) {
      const float4* r4 = reinterpret_cast<const float4*>(rw + (size_t)(h4 * 4 + j) * NEXP);
      float4 r0 = r4[0], r1 = r4[1];
      acc[0] += xs[j] * r0.x; acc[1] += xs[j] * r0.y;
      acc[2] += xs[j] * r0.z; acc[3] += xs[j] * r0.w;
      acc[4] += xs[j] * r1.x; acc[5] += xs[j] * r1.y;
      acc[6] += xs[j] * r1.z; acc[7] += xs[j] * r1.w;
    }
  }
  float mx = acc[0];
#pragma unroll
  for (int e = 1; e < NEXP; ++e) mx = fmaxf(mx, acc[e]);
  float p[NEXP];
#pragma unroll
  for (int e = 0; e < NEXP; ++e) p[e] = __expf(acc[e] - mx);
  // top-2 (ties -> lower index first, matching lax.top_k)
  int i1 = 0; float p1 = p[0];
#pragma unroll
  for (int e = 1; e < NEXP; ++e) if (p[e] > p1) { p1 = p[e]; i1 = e; }
  int i2 = -1; float p2 = -1.0f;
#pragma unroll
  for (int e = 0; e < NEXP; ++e) if (e != i1 && p[e] > p2) { p2 = p[e]; i2 = e; }
  float inv = 1.0f / (p1 + p2);    // softmax denom cancels in normalized top-2
  int s1 = atomicAdd(&cnt[i1], 1);
  ids[i1 * T_TOK + s1] = 2 * t;
  awts[2 * t] = p1 * inv;
  int s2 = atomicAdd(&cnt[i2], 1);
  ids[i2 * T_TOK + s2] = 2 * t + 1;
  awts[2 * t + 1] = p2 * inv;
}

// ---------------------------------------------------------------- x -> bf16
__global__ __launch_bounds__(256)
void xbf_kernel(const float* __restrict__ x, unsigned short* __restrict__ xbf)
{
  int i = blockIdx.x * 256 + threadIdx.x;        // one 8-elem chunk per thread
  if (i >= T_TOK * (H_DIM / 8)) return;
  const float4* src = reinterpret_cast<const float4*>(x) + (size_t)i * 2;
  float4 a = src[0], b = src[1];
  Frag f;
  f.s[0] = f2bf(a.x); f.s[1] = f2bf(a.y); f.s[2] = f2bf(a.z); f.s[3] = f2bf(a.w);
  f.s[4] = f2bf(b.x); f.s[5] = f2bf(b.y); f.s[6] = f2bf(b.z); f.s[7] = f2bf(b.w);
  reinterpret_cast<v4u*>(xbf)[i] = f.q[0];
}

// ---------------------------------------------------------------- weight swizzle
// in: per-expert [Kd][Nd] f32 (row-major, K-major) ->
// out: WMMA-B fragment order: group g=(e*(Nd/16)+nt)*(Kd/32)+kc, then
//      [lane][16 bf16] where lane={g2=lane>>4, n=lane&15}, half h -> K=kc*32+8*g2+h+(h>=8?8:0).
// A wave's fragment load becomes 1KB fully-contiguous (2 x b128 per lane).
__global__ __launch_bounds__(256)
void swizzle_kernel(const float* __restrict__ in, unsigned short* __restrict__ out,
                    int Kd, int Nd)
{
  const int lane = threadIdx.x & 31;
  const int wave = threadIdx.x >> 5;
  const int gpe = (Nd / 16) * (Kd / 32);
  const long long gid = (long long)blockIdx.x * 8 + wave;
  const int e  = (int)(gid / gpe);
  const int rem = (int)(gid % gpe);
  const int nt = rem / (Kd / 32);
  const int kc = rem % (Kd / 32);
  const int g2 = lane >> 4, ln = lane & 15;
  const float* inp = in + (size_t)e * Kd * Nd + (size_t)(kc * 32 + 8 * g2) * Nd + nt * 16 + ln;
  Frag f;
#pragma unroll
  for (int h = 0; h < 16; ++h) f.s[h] = f2bf(inp[(size_t)(h + (h >= 8 ? 8 : 0)) * Nd]);
  v4u* op = reinterpret_cast<v4u*>(out + (size_t)gid * FRAG_ELEMS + lane * 16);
  op[0] = f.q[0];
  op[1] = f.q[1];
}

// ---------------------------------------------------------------- fused expert FFN
// block = 8 waves (256 thr), one (expert, 32-token tile).
// y[32,1024] = gelu(X[32,1024] @ W1 + b1) @ W2 + b2, streamed over F in chunks of 64.
__global__ __launch_bounds__(256)
void moe_ffn_kernel(const unsigned short* __restrict__ xbf,   // [T][H] bf16
                    const unsigned short* __restrict__ w1s,   // frag-swizzled, N=F K=H
                    const unsigned short* __restrict__ w2s,   // frag-swizzled, N=H K=F
                    const float* __restrict__ b1,             // [E][F]
                    const float* __restrict__ b2,             // [E][H]
                    const int* __restrict__ cnt,
                    const int* __restrict__ ids,              // [E][T] assignment ids (2t+k)
                    float* __restrict__ ys)                   // [2T][H]
{
  const int tilesPerExpert = T_TOK / M_TILE;   // 256
  const int e    = blockIdx.x / tilesPerExpert;
  const int tile = blockIdx.x % tilesPerExpert;
  const int base = tile * M_TILE;
  const int ne   = cnt[e];
  if (base >= ne) return;

  __shared__ unsigned short Xs[M_TILE][XS_STRIDE];   // 65 KB  bf16 activations
  __shared__ unsigned short Hs[M_TILE][HS_STRIDE];   // 4.5 KB bf16 gelu(h) chunk
  __shared__ int Aid[M_TILE];
  __shared__ int Xrow[M_TILE];

  const int tid  = threadIdx.x;
  const int lane = tid & 31;
  const int wave = tid >> 5;   // 0..7
  const int wm   = wave >> 2;  // 0..1  (row tile)
  const int wn   = wave & 3;   // 0..3  (col quarter)
  const int g    = lane >> 4;  // lane group per ISA fragment layout
  const int ln   = lane & 15;

  if (tid < M_TILE) {
    int slot = base + tid;
    int a = (slot < ne) ? ids[e * T_TOK + slot] : -1;
    Aid[tid] = a;
    Xrow[tid] = (a >= 0) ? (a >> 1) : 0;
  }
  __syncthreads();

  // gather X tile (bf16, 16B chunks, coalesced). 4096 chunks over 256 threads.
#pragma unroll
  for (int p = 0; p < 16; ++p) {
    int chunk = tid + 256 * p;
    int row = chunk >> 7;            // 128 chunks per row
    int c8  = (chunk & 127) << 3;    // bf16 element offset
    const unsigned short* src = xbf + (size_t)Xrow[row] * H_DIM + c8;
#if defined(HAVE_ASYNC_LDS)
    __builtin_amdgcn_global_load_async_to_lds_b128(
        (gv4p)(vsi4*)src, (lv4p)(vsi4*)&Xs[row][c8], 0, 0);
#else
    *reinterpret_cast<v4u*>(&Xs[row][c8]) = *reinterpret_cast<const v4u*>(src);
#endif
  }
#if defined(HAVE_ASYNC_LDS)
#if __has_builtin(__builtin_amdgcn_s_wait_asynccnt)
  __builtin_amdgcn_s_wait_asynccnt(0);
#else
  asm volatile("s_wait_asynccnt 0" ::: "memory");
#endif
#endif
  __syncthreads();

  v8f zero = 0.0f;
  v8f acc[16];                       // this wave's y cols: wn*256 .. wn*256+255
#pragma unroll
  for (int i = 0; i < 16; ++i) acc[i] = zero;

  const int am = wm * 16 + ln;       // A-fragment row (lane&15 per ISA layout)
  // per-lane byte base inside each 1KB fragment
  const unsigned short* w2base =
      w2s + ((size_t)(e * (H_DIM / 16) + wn * 16) * (F_DIM / 32)) * FRAG_ELEMS + lane * 16;

  for (int fc = 0; fc < F_DIM; fc += FC) {
    // ---- GEMM1: h tile rows wm*16.., cols fc + wn*16 .. (+16), K = H
    v8f hacc = zero;
    const unsigned short* w1g =
        w1s + ((size_t)(e * (F_DIM / 16) + (fc >> 4) + wn) * (H_DIM / 32)) * FRAG_ELEMS
            + lane * 16;
#pragma unroll 4
    for (int kc32 = 0; kc32 < H_DIM / 32; ++kc32) {
      Frag a, b;
      int kc = kc32 * 32;
      a.q[0] = *reinterpret_cast<const v4u*>(&Xs[am][kc + 8 * g]);
      a.q[1] = *reinterpret_cast<const v4u*>(&Xs[am][kc + 16 + 8 * g]);
      const v4u* bp = reinterpret_cast<const v4u*>(w1g + (size_t)kc32 * FRAG_ELEMS);
      b.q[0] = bp[0];
      b.q[1] = bp[1];
      hacc = wmma_bf16(a.v, b.v, hacc);
    }
    const int hn = fc + wn * 16 + ln;   // F column for this lane
    float bias = b1[e * F_DIM + hn];
#pragma unroll
    for (int r = 0; r < 8; ++r) {
      float hv = gelu_tanh(hacc[r] + bias);
      Hs[wm * 16 + r + 8 * g][wn * 16 + ln] = f2bf(hv);
    }
    __syncthreads();

    // ---- GEMM2: acc += Hs[32,FC] @ W2[fc:fc+FC, wave's 256 cols]
#pragma unroll
    for (int kk32 = 0; kk32 < FC / 32; ++kk32) {
      Frag a;
      int kk = kk32 * 32;
      a.q[0] = *reinterpret_cast<const v4u*>(&Hs[am][kk + 8 * g]);
      a.q[1] = *reinterpret_cast<const v4u*>(&Hs[am][kk + 16 + 8 * g]);
#pragma unroll
      for (int nt = 0; nt < 16; ++nt) {
        const v4u* bp = reinterpret_cast<const v4u*>(
            w2base + ((size_t)(nt * (F_DIM / 32) + (fc >> 5) + kk32)) * FRAG_ELEMS);
        Frag b;
        b.q[0] = bp[0];
        b.q[1] = bp[1];
        acc[nt] = wmma_bf16(a.v, b.v, acc[nt]);
      }
    }
    __syncthreads();   // protect Hs before next chunk overwrites it
  }

  // ---- store per-assignment outputs (+b2); weight applied in combine kernel
  const float* b2e = b2 + e * H_DIM;
#pragma unroll
  for (int nt = 0; nt < 16; ++nt) {
    int ncol = wn * 256 + nt * 16 + ln;
    float bv = b2e[ncol];
#pragma unroll
    for (int r = 0; r < 8; ++r) {
      int m = wm * 16 + r + 8 * g;     // D layout: row = vgpr + 8*(lane>=16)
      int aid = Aid[m];
      if (aid >= 0) ys[(size_t)aid * H_DIM + ncol] = acc[nt][r] + bv;
    }
  }
}

// ---------------------------------------------------------------- combine
__global__ __launch_bounds__(256)
void combine_kernel(const float* __restrict__ ys, const float* __restrict__ awts,
                    float* __restrict__ out)
{
  int i = blockIdx.x * 256 + threadIdx.x;          // over T*H/4 float4s
  if (i >= T_TOK * (H_DIM / 4)) return;
  int t  = i >> 8;
  int c4 = (i & 255);
  float w0 = awts[2 * t], w1 = awts[2 * t + 1];
  float4 y0 = reinterpret_cast<const float4*>(ys + (size_t)(2 * t) * H_DIM)[c4];
  float4 y1 = reinterpret_cast<const float4*>(ys + (size_t)(2 * t + 1) * H_DIM)[c4];
  float4 o;
  o.x = w0 * y0.x + w1 * y1.x;
  o.y = w0 * y0.y + w1 * y1.y;
  o.z = w0 * y0.z + w1 * y1.z;
  o.w = w0 * y0.w + w1 * y1.w;
  reinterpret_cast<float4*>(out)[i] = o;
}

// ---------------------------------------------------------------- launcher
extern "C" void kernel_launch(void* const* d_in, const int* in_sizes, int n_in,
                              void* d_out, int out_size, void* d_ws, size_t ws_size,
                              hipStream_t stream)
{
  const float* x  = (const float*)d_in[0];
  const float* rw = (const float*)d_in[1];
  const float* rb = (const float*)d_in[2];
  const float* w1 = (const float*)d_in[3];
  const float* b1 = (const float*)d_in[4];
  const float* w2 = (const float*)d_in[5];
  const float* b2 = (const float*)d_in[6];
  // d_in[7] = top_k (fixed 2)
  float* out = (float*)d_out;

  char* ws = (char*)d_ws;
  size_t off = 0;
  int* cnt = (int*)(ws + off);                 off += 256;
  int* ids = (int*)(ws + off);                 off += (size_t)NEXP * T_TOK * sizeof(int);
  float* awts = (float*)(ws + off);            off += (size_t)2 * T_TOK * sizeof(float);
  unsigned short* w1s = (unsigned short*)(ws + off); off += (size_t)NEXP * F_DIM * H_DIM * 2;
  unsigned short* w2s = (unsigned short*)(ws + off); off += (size_t)NEXP * H_DIM * F_DIM * 2;
  unsigned short* xbf = (unsigned short*)(ws + off); off += (size_t)T_TOK * H_DIM * 2;
  float* ysbuf = (float*)(ws + off);           off += (size_t)2 * T_TOK * H_DIM * sizeof(float);
  (void)ws_size; (void)in_sizes; (void)n_in; (void)out_size;

  zero_cnt_kernel<<<1, 32, 0, stream>>>(cnt);
  router_kernel<<<T_TOK / 256, 256, 0, stream>>>(x, rw, rb, cnt, ids, awts);
  // weight fragment swizzle: 65536 fragment groups each, 8 per block
  swizzle_kernel<<<(NEXP * (F_DIM / 16) * (H_DIM / 32)) / 8, 256, 0, stream>>>(w1, w1s, H_DIM, F_DIM);
  swizzle_kernel<<<(NEXP * (H_DIM / 16) * (F_DIM / 32)) / 8, 256, 0, stream>>>(w2, w2s, F_DIM, H_DIM);
  xbf_kernel<<<(T_TOK * (H_DIM / 8)) / 256, 256, 0, stream>>>(x, xbf);
  moe_ffn_kernel<<<NEXP * (T_TOK / M_TILE), 256, 0, stream>>>(xbf, w1s, w2s, b1, b2, cnt, ids, ysbuf);
  combine_kernel<<<(T_TOK * (H_DIM / 4)) / 256, 256, 0, stream>>>(ysbuf, awts, out);
}
